// DGNRNetwork_72155450573513
// MI455X (gfx1250) — matrix operand order repaired and checked by reading, not compiled, not run on hardware
//
#include <hip/hip_runtime.h>
#include <hip/hip_bf16.h>
#include <cstdint>
#include <cstddef>

// Problem constants (from the reference)
#define BSZ   32
#define NPB   1000
#define NODES (BSZ * NPB)     // 32000
#define DIMF  6
#define HIDN  64
#define NH    4
#define CHN   (NH * HIDN)     // 256
#define OUTD  5
#define OBSROW (NPB * (2 + DIMF) + 1) // 8001

typedef __attribute__((ext_vector_type(16))) _Float16 v16h;
typedef __attribute__((ext_vector_type(8)))  _Float16 v8h;
typedef __attribute__((ext_vector_type(2)))  _Float16 v2h;
typedef __attribute__((ext_vector_type(8)))  float    v8f;

// ---------------------------------------------------------------------------
// Utility fills
// ---------------------------------------------------------------------------
__global__ __launch_bounds__(256)
void fill_i32_kernel(int* __restrict__ p, long long n, int v) {
    long long t = (long long)blockIdx.x * blockDim.x + threadIdx.x;
    if (t < n) p[t] = v;
}

// ---------------------------------------------------------------------------
// Encoder: feats(6) -> relu(W1) -> relu(W2) ; outputs f16 x1 (NODES x 64)
// ---------------------------------------------------------------------------
__global__ __launch_bounds__(256)
void enc1_kernel(const float* __restrict__ obs,
                 const float* __restrict__ w1, const float* __restrict__ b1,
                 float* __restrict__ h1) {
    int t = blockIdx.x * blockDim.x + threadIdx.x;
    if (t >= NODES * HIDN) return;
    int g = t >> 6, j = t & 63;
    int b = g / NPB, nn = g % NPB;
    const float* nd = obs + (size_t)b * OBSROW + (size_t)nn * (2 + DIMF);
    float acc = b1[j];
#pragma unroll
    for (int i = 0; i < DIMF; ++i) acc += nd[2 + i] * w1[i * HIDN + j];
    h1[t] = fmaxf(acc, 0.0f);
}

__global__ __launch_bounds__(256)
void enc2_kernel(const float* __restrict__ h1,
                 const float* __restrict__ w2, const float* __restrict__ b2,
                 _Float16* __restrict__ x1) {
    int t = blockIdx.x * blockDim.x + threadIdx.x;
    if (t >= NODES * HIDN) return;
    int g = t >> 6, j = t & 63;
    const float* hr = h1 + (size_t)g * HIDN;
    float acc = b2[j];
#pragma unroll 8
    for (int i = 0; i < HIDN; ++i) acc += hr[i] * w2[i * HIDN + j];
    x1[t] = (_Float16)fmaxf(acc, 0.0f);
}

// ---------------------------------------------------------------------------
// Weight convert + transpose: w[K][256] f32  ->  wT[256][K] f16
// ---------------------------------------------------------------------------
__global__ __launch_bounds__(256)
void wconv_kernel(const float* __restrict__ w, _Float16* __restrict__ wT, int K) {
    int t = blockIdx.x * blockDim.x + threadIdx.x;
    if (t >= K * CHN) return;
    int n = t / K, k = t % K;
    wT[t] = (_Float16)w[(size_t)k * CHN + n];
}

// ---------------------------------------------------------------------------
// QKV GEMM via v_wmma_f32_16x16x32_f16.
// One wave computes a 16x64 strip (4 col-tiles) of one of {Q,K,V}:
// the A fragment is loaded once per K-step and reused by 4 independent WMMAs
// (hides XDL latency, quarters activation re-reads).
// A fragment: lane(0-15)=row m, half-wave selects K-subgroup; two b128 loads.
// B fragment: wT is [N][K] so each lane (=column) loads contiguous K; two b128.
// D: lane column = lane%16, rows = (lane/16)*8 + r.
// Grid: (NODES/16) * (16/4) * 3 = 24000 waves = 3000 blocks x 8 waves (exact).
// ---------------------------------------------------------------------------
__global__ __launch_bounds__(256)
void qkv_wmma_kernel(const _Float16* __restrict__ x, int K,
                     const _Float16* __restrict__ wTq,
                     const _Float16* __restrict__ wTk,
                     const _Float16* __restrict__ wTv,
                     const float* __restrict__ bq,
                     const float* __restrict__ bk,
                     const float* __restrict__ bv,
                     _Float16* __restrict__ qo,
                     _Float16* __restrict__ ko,
                     _Float16* __restrict__ vo) {
    const int ROW_TILES = NODES / 16;     // 2000
    const int COL_GRPS  = (CHN / 16) / 4; // 4 groups of 4 col-tiles
    int wave = blockIdx.x * (blockDim.x >> 5) + (threadIdx.x >> 5);
    int total = ROW_TILES * COL_GRPS * 3;
    if (wave >= total) return;  // whole-wave divergence only

    int mat = wave / (ROW_TILES * COL_GRPS);
    int rem = wave % (ROW_TILES * COL_GRPS);
    int rt = rem / COL_GRPS;
    int cg = rem % COL_GRPS;

    const _Float16* wT   = (mat == 0) ? wTq : (mat == 1) ? wTk : wTv;
    const float*    bias = (mat == 0) ? bq  : (mat == 1) ? bk  : bv;
    _Float16*       outp = (mat == 0) ? qo  : (mat == 1) ? ko  : vo;

    int lane = threadIdx.x & 31;
    int half = lane >> 4;
    int l16  = lane & 15;

    const _Float16* xrow = x + (size_t)(rt * 16 + l16) * K;
    const _Float16* wrow0 = wT + (size_t)(cg * 64 + l16) * K;

    v8f c0 = {}, c1 = {}, c2 = {}, c3 = {};
    for (int k0 = 0; k0 < K; k0 += 32) {
        v8h a0 = *(const v8h*)(xrow + k0 + half * 16);
        v8h a1 = *(const v8h*)(xrow + k0 + half * 16 + 8);
        v16h A;
#pragma unroll
        for (int i = 0; i < 8; ++i) { A[i] = a0[i]; A[i + 8] = a1[i]; }
#pragma unroll
        for (int j = 0; j < 4; ++j) {
            const _Float16* wrow = wrow0 + (size_t)(j * 16) * K;
            v8h b0 = *(const v8h*)(wrow + k0 + half * 16);
            v8h b1 = *(const v8h*)(wrow + k0 + half * 16 + 8);
            v16h B;
#pragma unroll
            for (int i = 0; i < 8; ++i) { B[i] = b0[i]; B[i + 8] = b1[i]; }
            v8f& c = (j == 0) ? c0 : (j == 1) ? c1 : (j == 2) ? c2 : c3;
            c = __builtin_amdgcn_wmma_f32_16x16x32_f16(
                /*neg_a=*/false, A, /*neg_b=*/false, B,
                /*c_mod=*/(short)0, c, /*reuse_a=*/false, /*reuse_b=*/false);
        }
    }

#pragma unroll
    for (int j = 0; j < 4; ++j) {
        v8f& c = (j == 0) ? c0 : (j == 1) ? c1 : (j == 2) ? c2 : c3;
        int col = cg * 64 + j * 16 + l16;
        float bcol = bias[col];
        _Float16* orow = outp + (size_t)(rt * 16 + half * 8) * CHN + col;
#pragma unroll
        for (int r = 0; r < 8; ++r)
            orow[(size_t)r * CHN] = (_Float16)(c[r] + bcol);
    }
}

// ---------------------------------------------------------------------------
// CSR-by-destination build (runs once, reused by both layers).
// ---------------------------------------------------------------------------
__global__ __launch_bounds__(256)
void count_kernel(const long long* __restrict__ dst, int E, int* __restrict__ cnt) {
    int e = blockIdx.x * blockDim.x + threadIdx.x;
    if (e >= E) return;
    atomicAdd(cnt + (int)dst[e], 1);
}

// Exclusive scan of cnt[NODES] -> rowptr[NODES+1]; single block, LDS partials.
__global__ __launch_bounds__(256)
void scan_kernel(const int* __restrict__ cnt, int* __restrict__ rowptr) {
    __shared__ int partial[256];
    const int CHUNK = NODES / 256;  // 125, exact
    int tid = threadIdx.x;
    int start = tid * CHUNK;
    int sum = 0;
    for (int i = 0; i < CHUNK; ++i) sum += cnt[start + i];
    partial[tid] = sum;
    __syncthreads();
    if (tid == 0) {
        int run = 0;
        for (int i = 0; i < 256; ++i) { int t = partial[i]; partial[i] = run; run += t; }
        rowptr[NODES] = run;
    }
    __syncthreads();
    int run = partial[tid];
    for (int i = 0; i < CHUNK; ++i) { rowptr[start + i] = run; run += cnt[start + i]; }
}

__global__ __launch_bounds__(256)
void scatter_kernel(const long long* __restrict__ src, const long long* __restrict__ dst,
                    int E, const int* __restrict__ rowptr, int* __restrict__ fill,
                    int* __restrict__ adj) {
    int e = blockIdx.x * blockDim.x + threadIdx.x;
    if (e >= E) return;
    int d = (int)dst[e];
    int pos = rowptr[d] + atomicAdd(fill + d, 1);
    adj[pos] = (int)src[e];
}

// ---------------------------------------------------------------------------
// Fused gather aggregation: one wave per (node, head); 2 channels per lane.
// Online softmax (running max / sum / weighted accum) entirely in registers,
// per-edge score reduced across the wave with shfl_xor. Fuses ReLU + f16 out.
// No float atomics anywhere.
// ---------------------------------------------------------------------------
__global__ __launch_bounds__(256)
void agg_kernel(const int* __restrict__ rowptr, const int* __restrict__ adj,
                const _Float16* __restrict__ q, const _Float16* __restrict__ k,
                const _Float16* __restrict__ v, _Float16* __restrict__ xo) {
    int wave = blockIdx.x * (blockDim.x >> 5) + (threadIdx.x >> 5);
    if (wave >= NODES * NH) return;
    int node = wave >> 2, h = wave & 3;
    int lane = threadIdx.x & 31;

    int beg = rowptr[node], end = rowptr[node + 1];

    const size_t chan = (size_t)h * HIDN + lane * 2;
    v2h qv = *(const v2h*)(q + (size_t)node * CHN + chan);
    float q0 = (float)qv[0], q1 = (float)qv[1];

    float m = -INFINITY, z = 0.0f, a0 = 0.0f, a1 = 0.0f;
    for (int p = beg; p < end; ++p) {
        int s = adj[p];
        v2h kv = *(const v2h*)(k + (size_t)s * CHN + chan);
        v2h vv = *(const v2h*)(v + (size_t)s * CHN + chan);
        float part = q0 * (float)kv[0] + q1 * (float)kv[1];
#pragma unroll
        for (int off = 16; off > 0; off >>= 1)
            part += __shfl_xor(part, off, 32);
        float sc = part * 0.125f;                  // 1/sqrt(64)
        float mn = fmaxf(m, sc);
        float rescale = __expf(m - mn);            // 0 on first edge (m=-inf)
        float ee = __expf(sc - mn);
        z  = z  * rescale + ee;
        a0 = a0 * rescale + ee * (float)vv[0];
        a1 = a1 * rescale + ee * (float)vv[1];
        m = mn;
    }
    float inv = (end > beg) ? (1.0f / z) : 0.0f;   // isolated node -> 0 (matches segment_sum)
    v2h o;
    o[0] = (_Float16)fmaxf(a0 * inv, 0.0f);
    o[1] = (_Float16)fmaxf(a1 * inv, 0.0f);
    *(v2h*)(xo + (size_t)node * CHN + chan) = o;
}

// ---------------------------------------------------------------------------
// Head: gather agent rows and apply (256 -> 5) linear.
// ---------------------------------------------------------------------------
__global__ __launch_bounds__(256)
void head_kernel(const float* __restrict__ obs, const _Float16* __restrict__ h,
                 const float* __restrict__ ow, const float* __restrict__ ob,
                 float* __restrict__ out) {
    int t = blockIdx.x * blockDim.x + threadIdx.x;
    if (t >= BSZ * OUTD) return;
    int b = t / OUTD, o = t % OUTD;
    float af = obs[(size_t)b * OBSROW + (OBSROW - 1)];
    int ag = (int)fminf(fmaxf(af, 0.0f), (float)(NPB - 1));
    const _Float16* hr = h + ((size_t)b * NPB + ag) * CHN;
    float acc = ob[o];
#pragma unroll 8
    for (int i = 0; i < CHN; ++i) acc += (float)hr[i] * ow[i * OUTD + o];
    out[t] = acc;
}

// ---------------------------------------------------------------------------
// Launcher
// ---------------------------------------------------------------------------
extern "C" void kernel_launch(void* const* d_in, const int* in_sizes, int n_in,
                              void* d_out, int out_size, void* d_ws, size_t ws_size,
                              hipStream_t stream) {
    const float*     obs  = (const float*)d_in[0];
    const long long* ei   = (const long long*)d_in[1];
    const float* enc_w1 = (const float*)d_in[2];
    const float* enc_b1 = (const float*)d_in[3];
    const float* enc_w2 = (const float*)d_in[4];
    const float* enc_b2 = (const float*)d_in[5];
    const float* c1_wq = (const float*)d_in[6],  *c1_bq = (const float*)d_in[7];
    const float* c1_wk = (const float*)d_in[8],  *c1_bk = (const float*)d_in[9];
    const float* c1_wv = (const float*)d_in[10], *c1_bv = (const float*)d_in[11];
    const float* c2_wq = (const float*)d_in[12], *c2_bq = (const float*)d_in[13];
    const float* c2_wk = (const float*)d_in[14], *c2_bk = (const float*)d_in[15];
    const float* c2_wv = (const float*)d_in[16], *c2_bv = (const float*)d_in[17];
    const float* out_w = (const float*)d_in[18], *out_b = (const float*)d_in[19];
    float* out = (float*)d_out;

    const int E = in_sizes[1] / 2;
    const long long* src = ei;
    const long long* dst = ei + E;

    // ---- workspace carve-up (256-byte aligned) ----
    char* ws = (char*)d_ws;
    size_t off = 0;
    auto take = [&](size_t bytes) -> size_t {
        size_t o = off;
        off += (bytes + 255) & ~(size_t)255;
        return o;
    };
    _Float16* x1  = (_Float16*)(ws + take((size_t)NODES * HIDN * sizeof(_Float16)));
    _Float16* x2  = (_Float16*)(ws + take((size_t)NODES * CHN  * sizeof(_Float16)));
    _Float16* x3  = (_Float16*)(ws + take((size_t)NODES * CHN  * sizeof(_Float16)));
    _Float16* wTq = (_Float16*)(ws + take((size_t)CHN * CHN * sizeof(_Float16)));
    _Float16* wTk = (_Float16*)(ws + take((size_t)CHN * CHN * sizeof(_Float16)));
    _Float16* wTv = (_Float16*)(ws + take((size_t)CHN * CHN * sizeof(_Float16)));
    _Float16* qb  = (_Float16*)(ws + take((size_t)NODES * CHN * sizeof(_Float16)));
    _Float16* kb  = (_Float16*)(ws + take((size_t)NODES * CHN * sizeof(_Float16)));
    _Float16* vb  = (_Float16*)(ws + take((size_t)NODES * CHN * sizeof(_Float16)));
    float*    h1  = (float*)(ws + take((size_t)NODES * HIDN * sizeof(float)));
    int*      cnt = (int*)(ws + take((size_t)NODES * sizeof(int)));
    int*      fil = (int*)(ws + take((size_t)NODES * sizeof(int)));
    int*      rp  = (int*)(ws + take((size_t)(NODES + 1) * sizeof(int)));
    int*      adj = (int*)(ws + take((size_t)(E > 0 ? E : 1) * sizeof(int)));
    (void)ws_size; (void)n_in; (void)out_size;

    const int T = 256;
    auto blk = [](long long n, int t) { return (unsigned)((n + t - 1) / t); };

    const int QKV_BLOCKS = (NODES / 16) * ((CHN / 16) / 4) * 3 / 8;  // 3000, exact
    const int AGG_BLOCKS = NODES * NH / 8;                           // 16000, exact

    // ---- encoder ----
    enc1_kernel<<<blk((long long)NODES * HIDN, T), T, 0, stream>>>(obs, enc_w1, enc_b1, h1);
    enc2_kernel<<<blk((long long)NODES * HIDN, T), T, 0, stream>>>(h1, enc_w2, enc_b2, x1);

    // ---- CSR by destination (built once, reused by both layers) ----
    fill_i32_kernel<<<blk(NODES, T), T, 0, stream>>>(cnt, NODES, 0);
    fill_i32_kernel<<<blk(NODES, T), T, 0, stream>>>(fil, NODES, 0);
    if (E > 0)
        count_kernel<<<blk(E, T), T, 0, stream>>>(dst, E, cnt);
    scan_kernel<<<1, T, 0, stream>>>(cnt, rp);
    if (E > 0)
        scatter_kernel<<<blk(E, T), T, 0, stream>>>(src, dst, E, rp, fil, adj);

    // ---- layer 1 ----
    wconv_kernel<<<blk((long long)HIDN * CHN, T), T, 0, stream>>>(c1_wq, wTq, HIDN);
    wconv_kernel<<<blk((long long)HIDN * CHN, T), T, 0, stream>>>(c1_wk, wTk, HIDN);
    wconv_kernel<<<blk((long long)HIDN * CHN, T), T, 0, stream>>>(c1_wv, wTv, HIDN);
    qkv_wmma_kernel<<<QKV_BLOCKS, T, 0, stream>>>(x1, HIDN, wTq, wTk, wTv,
                                                  c1_bq, c1_bk, c1_bv, qb, kb, vb);
    agg_kernel<<<AGG_BLOCKS, T, 0, stream>>>(rp, adj, qb, kb, vb, x2);

    // ---- layer 2 ----
    wconv_kernel<<<blk((long long)CHN * CHN, T), T, 0, stream>>>(c2_wq, wTq, CHN);
    wconv_kernel<<<blk((long long)CHN * CHN, T), T, 0, stream>>>(c2_wk, wTk, CHN);
    wconv_kernel<<<blk((long long)CHN * CHN, T), T, 0, stream>>>(c2_wv, wTv, CHN);
    qkv_wmma_kernel<<<QKV_BLOCKS, T, 0, stream>>>(x2, CHN, wTq, wTk, wTv,
                                                  c2_bq, c2_bk, c2_bv, qb, kb, vb);
    agg_kernel<<<AGG_BLOCKS, T, 0, stream>>>(rp, adj, qb, kb, vb, x3);

    // ---- head ----
    head_kernel<<<1, T, 0, stream>>>(obs, x3, out_w, out_b, out);
}
